// GroupQueryAttention_25237227832040
// MI455X (gfx1250) — compile-verified, matrix operand order checked by reading, not run on hardware
//
#include <hip/hip_runtime.h>
#include <hip/hip_bf16.h>
#include <stdint.h>

typedef __bf16 bf16_t;
typedef __attribute__((ext_vector_type(16))) __bf16 v16bf;
typedef __attribute__((ext_vector_type(8)))  float  v8f;
typedef __attribute__((ext_vector_type(4)))  unsigned u32x4;
typedef __attribute__((ext_vector_type(8)))  int      i32x8;
typedef __attribute__((ext_vector_type(4)))  int      i32x4;

#define B_   2
#define S_   2048
#define D_   2048
#define KD_  512
#define NH_  32
#define G_   8
#define HD_  64

struct U4x2 { uint4 a, b; };

static __device__ __forceinline__ v16bf make_frag(uint4 lo, uint4 hi) {
  U4x2 u; u.a = lo; u.b = hi;
  return __builtin_bit_cast(v16bf, u);
}

static __device__ __forceinline__ v8f wmma_bf16(v16bf a, v16bf b, v8f c) {
  // D = A*B + C, 16x16x32 bf16 -> f32
  return __builtin_amdgcn_wmma_f32_16x16x32_bf16(false, a, false, b, (short)0, c, false, false);
}

// LDS byte offset of a __shared__ object (addrspace(3) ptrtoint, NOT flat addr)
static __device__ __forceinline__ unsigned lds_off_of(void* p) {
  return (unsigned)(unsigned long long)(__attribute__((address_space(3))) void*)p;
}

// Tensor Data Mover: DMA a 2D tile (rows x 32 bf16 elems, row stride = stride elems)
// from global memory into LDS at lds_off. D# built per CDNA5 ISA ch.8.
static __device__ __forceinline__ void tdm_load_2d(unsigned lds_off, const bf16_t* g,
                                                   unsigned rows, unsigned stride) {
  unsigned long long ga = (unsigned long long)g;
  u32x4 g0;
  g0[0] = 1u;                                   // count=1 (valid), user mode
  g0[1] = lds_off;                              // lds_addr (bytes)
  g0[2] = (unsigned)ga;                         // global_addr[31:0]
  g0[3] = (unsigned)((ga >> 32) & 0x1ffffffu)   // global_addr[56:32]
        | (2u << 30);                           // type=2 ("image")
  i32x8 g1;
  g1[0] = (int)(1u << 16);                      // workgroup_mask=0, data_size=1 (2B)
  g1[1] = (int)((stride & 0xffffu) << 16);      // tensor_dim0[15:0]  (=stride)
  g1[2] = (int)(((stride >> 16) & 0xffffu)      // tensor_dim0[31:16]
        | ((rows & 0xffffu) << 16));            // tensor_dim1[15:0]  (=rows)
  g1[3] = (int)(((rows >> 16) & 0xffffu)        // tensor_dim1[31:16]
        | (32u << 16));                         // tile_dim0 = 32 elems
  g1[4] = (int)(rows & 0xffffu);                // tile_dim1 = rows, tile_dim2 = 0
  g1[5] = (int)stride;                          // tensor_dim0_stride[31:0]
  g1[6] = 0;                                    // stride[47:32]=0, dim1_stride lo=0
  g1[7] = 0;
  i32x4 zz4; zz4[0] = 0; zz4[1] = 0; zz4[2] = 0; zz4[3] = 0;
  i32x8 zz8;
#pragma unroll
  for (int i = 0; i < 8; ++i) zz8[i] = 0;
  // clang-23 / therock-10.0 form: (g0, g1, g2, g3, g4, cpol)
  __builtin_amdgcn_tensor_load_to_lds(g0, g1, zz4, zz4, zz8, 0);
}

// ---------------- elementwise cast f32 -> bf16 ----------------
__global__ __launch_bounds__(256) void cast_kernel(const float* __restrict__ src,
                                                   bf16_t* __restrict__ dst, int n) {
  int i = (blockIdx.x * 256 + threadIdx.x) * 4;
  if (i + 3 < n) {
    float4 f = *(const float4*)(src + i);
    dst[i + 0] = (bf16_t)f.x;
    dst[i + 1] = (bf16_t)f.y;
    dst[i + 2] = (bf16_t)f.z;
    dst[i + 3] = (bf16_t)f.w;
  }
}

// ------------- transpose+cast: src f32 [R][C] -> dst bf16 [C][R] -------------
__global__ __launch_bounds__(256) void tcast_kernel(const float* __restrict__ src,
                                                    bf16_t* __restrict__ dst, int R, int C) {
  __shared__ float t[16][17];
  int nbx = C >> 4;
  int bx = blockIdx.x % nbx, by = blockIdx.x / nbx;
  int tx = threadIdx.x & 15, ty = threadIdx.x >> 4;
  t[ty][tx] = src[(size_t)(by * 16 + ty) * C + bx * 16 + tx];
  __syncthreads();
  dst[(size_t)(bx * 16 + ty) * R + by * 16 + tx] = (bf16_t)t[tx][ty];
}

// ------------- transpose bf16: src [R][C] -> dst [C][R], batched (y) -------------
__global__ __launch_bounds__(256) void tbf_kernel(const bf16_t* __restrict__ src,
                                                  bf16_t* __restrict__ dst, int R, int C) {
  __shared__ bf16_t t[16][17];
  int nbx = C >> 4;
  int bx = blockIdx.x % nbx, by = blockIdx.x / nbx;
  int tx = threadIdx.x & 15, ty = threadIdx.x >> 4;
  const bf16_t* s = src + (size_t)blockIdx.y * R * C;
  bf16_t*       d = dst + (size_t)blockIdx.y * R * C;
  t[ty][tx] = s[(size_t)(by * 16 + ty) * C + bx * 16 + tx];
  __syncthreads();
  d[(size_t)(bx * 16 + ty) * R + by * 16 + tx] = t[tx][ty];
}

// ------------- WMMA GEMM: C[M,N] = A[M,K](bf16) * Wt[N,K]^T(bf16) + bias -------------
// Block tile 256x128, 8 waves (4x2), each wave 64x64 = 4x4 wmma tiles, K-step 32.
// A/B tiles staged to LDS by the Tensor Data Mover, double buffered (TENSORcnt).
template <bool F32OUT>
__global__ __launch_bounds__(256) void gemm_kernel(const bf16_t* __restrict__ A,
                                                   const bf16_t* __restrict__ Wt,
                                                   const float* __restrict__ bias,
                                                   void* __restrict__ Cout,
                                                   int M, int N, int Kd) {
  __shared__ bf16_t As[2][256][32];  // 2 x 16 KB
  __shared__ bf16_t Bs[2][128][32];  // 2 x  8 KB
  const int tid  = threadIdx.x;
  const int wid  = tid >> 5, lane = tid & 31;
  const int hsel = lane >> 4, lrow = lane & 15;
  const int mBase = blockIdx.x * 256;
  const int nBase = blockIdx.y * 128;
  const int wm = wid & 3, wn = wid >> 2;  // 4 x 2 wave grid

  v8f acc[4][4] = {};

  // prologue: TDM fill buffer 0
  if (wid == 0) {
    tdm_load_2d(lds_off_of(&As[0][0][0]), A  + (size_t)mBase * Kd, 256, (unsigned)Kd);
    tdm_load_2d(lds_off_of(&Bs[0][0][0]), Wt + (size_t)nBase * Kd, 128, (unsigned)Kd);
  }

  for (int k0 = 0; k0 < Kd; k0 += 32) {
    const int buf = (k0 >> 5) & 1;
    if (wid == 0) {
      if (k0 + 32 < Kd) {
        // prefetch next k-tile into the other buffer, then wait for current pair
        tdm_load_2d(lds_off_of(&As[buf ^ 1][0][0]),
                    A + (size_t)mBase * Kd + (k0 + 32), 256, (unsigned)Kd);
        tdm_load_2d(lds_off_of(&Bs[buf ^ 1][0][0]),
                    Wt + (size_t)nBase * Kd + (k0 + 32), 128, (unsigned)Kd);
        __builtin_amdgcn_s_wait_tensorcnt(2);
      } else {
        __builtin_amdgcn_s_wait_tensorcnt(0);
      }
    }
    __syncthreads();

    v16bf af[4];
#pragma unroll
    for (int i = 0; i < 4; ++i) {
      const bf16_t* ar = &As[buf][wm * 64 + i * 16 + lrow][0];
      af[i] = make_frag(*(const uint4*)(ar + hsel * 8),
                        *(const uint4*)(ar + 16 + hsel * 8));
    }
#pragma unroll
    for (int j = 0; j < 4; ++j) {
      const bf16_t* br = &Bs[buf][wn * 64 + j * 16 + lrow][0];
      v16bf bfrag = make_frag(*(const uint4*)(br + hsel * 8),
                              *(const uint4*)(br + 16 + hsel * 8));
#pragma unroll
      for (int i = 0; i < 4; ++i)
        acc[i][j] = wmma_bf16(af[i], bfrag, acc[i][j]);
    }
    __syncthreads();  // all waves done reading buf before TDM overwrites it
  }

#pragma unroll
  for (int i = 0; i < 4; ++i) {
#pragma unroll
    for (int j = 0; j < 4; ++j) {
      int mt = mBase + wm * 64 + i * 16;
      int nt = nBase + wn * 64 + j * 16;
      int n  = nt + lrow;
      float bv = bias[n];
#pragma unroll
      for (int r = 0; r < 8; ++r) {
        int   m = mt + r + hsel * 8;  // C layout: vgpr r = rows r / r+8
        float v = acc[i][j][r] + bv;
        if (F32OUT) ((float*)Cout)[(size_t)m * N + n] = v;
        else        ((bf16_t*)Cout)[(size_t)m * N + n] = (bf16_t)v;
      }
    }
  }
}

// ------------- flash attention: one wave per (b, head, 16-query tile) -------------
__global__ __launch_bounds__(256) void attn_kernel(const bf16_t* __restrict__ Q,
                                                   const bf16_t* __restrict__ Kmat,
                                                   const bf16_t* __restrict__ Vt,
                                                   bf16_t* __restrict__ ctx) {
  __shared__ bf16_t Plds[8][16][32];  // per-wave P transpose staging
  const int wid  = threadIdx.x >> 5;
  const int lane = threadIdx.x & 31;
  const int hsel = lane >> 4;
  const int lrow = lane & 15;

  const int QT   = S_ / 16;
  int tile = blockIdx.x * 8 + wid;
  int qt = tile % QT;
  int h  = (tile / QT) % NH_;
  int b  = tile / (QT * NH_);
  int g  = h >> 2;  // 4 query heads per KV group

  // Q fragments (held in registers for the whole K loop)
  const bf16_t* qbase = Q + ((size_t)(b * S_ + qt * 16 + lrow)) * D_ + h * HD_;
  v16bf qa = make_frag(*(const uint4*)(qbase + hsel * 8),
                       *(const uint4*)(qbase + 16 + hsel * 8));
  v16bf qb = make_frag(*(const uint4*)(qbase + 32 + hsel * 8),
                       *(const uint4*)(qbase + 48 + hsel * 8));

  float rmax[8], rsum[8];
#pragma unroll
  for (int r = 0; r < 8; ++r) { rmax[r] = -1e30f; rsum[r] = 0.f; }
  v8f o[4] = {};

  const float scale = 0.125f;  // 1/sqrt(64)

  for (int kt = 0; kt < S_; kt += 32) {
    const bf16_t* kbaseA = Kmat + ((size_t)(b * S_ + kt + lrow)) * KD_ + g * HD_;
    const bf16_t* kbaseB = kbaseA + (size_t)16 * KD_;
    if (kt + 32 < S_) __builtin_prefetch(kbaseA + (size_t)32 * KD_, 0, 1);

    v16bf kA0 = make_frag(*(const uint4*)(kbaseA + hsel * 8),
                          *(const uint4*)(kbaseA + 16 + hsel * 8));
    v16bf kA1 = make_frag(*(const uint4*)(kbaseA + 32 + hsel * 8),
                          *(const uint4*)(kbaseA + 48 + hsel * 8));
    v16bf kB0 = make_frag(*(const uint4*)(kbaseB + hsel * 8),
                          *(const uint4*)(kbaseB + 16 + hsel * 8));
    v16bf kB1 = make_frag(*(const uint4*)(kbaseB + 32 + hsel * 8),
                          *(const uint4*)(kbaseB + 48 + hsel * 8));

    v8f z  = {};
    v8f sA = wmma_bf16(qa, kA0, z); sA = wmma_bf16(qb, kA1, sA);
    v8f sB = wmma_bf16(qa, kB0, z); sB = wmma_bf16(qb, kB1, sB);

    // V fragments: Vt is [b,g][64][S], contiguous along keys
    v16bf vf[4];
#pragma unroll
    for (int t = 0; t < 4; ++t) {
      const bf16_t* vb = Vt + ((size_t)((b * G_ + g) * HD_ + t * 16 + lrow)) * S_ + kt;
      vf[t] = make_frag(*(const uint4*)(vb + hsel * 8),
                        *(const uint4*)(vb + 16 + hsel * 8));
    }

    float corr[8];
#pragma unroll
    for (int r = 0; r < 8; ++r) {
      float sa = sA[r] * scale;
      float sb = sB[r] * scale;
      float mx = fmaxf(sa, sb);
      mx = fmaxf(mx, __shfl_xor(mx, 1, 32));
      mx = fmaxf(mx, __shfl_xor(mx, 2, 32));
      mx = fmaxf(mx, __shfl_xor(mx, 4, 32));
      mx = fmaxf(mx, __shfl_xor(mx, 8, 32));
      float mnew = fmaxf(rmax[r], mx);
      float pa = __expf(sa - mnew);
      float pb = __expf(sb - mnew);
      float ps = pa + pb;
      ps += __shfl_xor(ps, 1, 32);
      ps += __shfl_xor(ps, 2, 32);
      ps += __shfl_xor(ps, 4, 32);
      ps += __shfl_xor(ps, 8, 32);
      float c = __expf(rmax[r] - mnew);
      rsum[r] = rsum[r] * c + ps;
      rmax[r] = mnew;
      corr[r] = c;
      // C layout (row = r + hsel*8, key col = lrow / 16+lrow) -> A layout via LDS
      Plds[wid][r + hsel * 8][lrow]      = (bf16_t)pa;
      Plds[wid][r + hsel * 8][16 + lrow] = (bf16_t)pb;
    }
    asm volatile("s_wait_dscnt 0" ::: "memory");
    const bf16_t* prow = &Plds[wid][lrow][0];
    v16bf pf = make_frag(*(const uint4*)(prow + hsel * 8),
                         *(const uint4*)(prow + 16 + hsel * 8));
#pragma unroll
    for (int t = 0; t < 4; ++t) {
#pragma unroll
      for (int r = 0; r < 8; ++r) o[t][r] *= corr[r];
      o[t] = wmma_bf16(pf, vf[t], o[t]);
    }
  }

#pragma unroll
  for (int r = 0; r < 8; ++r) {
    float inv = 1.f / rsum[r];
    size_t row = (size_t)(b * S_ + qt * 16 + r + hsel * 8);
    bf16_t* cb = ctx + row * D_ + h * HD_ + lrow;
#pragma unroll
    for (int t = 0; t < 4; ++t) cb[t * 16] = (bf16_t)(o[t][r] * inv);
  }
}

// --------------------------------------------------------------------------
extern "C" void kernel_launch(void* const* d_in, const int* in_sizes, int n_in,
                              void* d_out, int out_size, void* d_ws, size_t ws_size,
                              hipStream_t stream) {
  const float* x  = (const float*)d_in[0];
  const float* wq = (const float*)d_in[1];
  const float* bq = (const float*)d_in[2];
  const float* wk = (const float*)d_in[3];
  const float* bk = (const float*)d_in[4];
  const float* wv = (const float*)d_in[5];
  const float* bv = (const float*)d_in[6];
  const float* wo = (const float*)d_in[7];
  const float* bo = (const float*)d_in[8];

  char* ws = (char*)d_ws;
  size_t off = 0;
  auto carve = [&](size_t bytes) {
    void* p = ws + off;
    off += (bytes + 255) & ~(size_t)255;
    return p;
  };
  const size_t M = (size_t)B_ * S_;  // 4096
  bf16_t* xb  = (bf16_t*)carve(M * D_ * 2);
  bf16_t* wqT = (bf16_t*)carve((size_t)D_ * D_ * 2);
  bf16_t* wkT = (bf16_t*)carve((size_t)D_ * KD_ * 2);
  bf16_t* wvT = (bf16_t*)carve((size_t)D_ * KD_ * 2);
  bf16_t* woT = (bf16_t*)carve((size_t)D_ * D_ * 2);
  bf16_t* Qb  = (bf16_t*)carve(M * D_ * 2);
  bf16_t* Kb  = (bf16_t*)carve(M * KD_ * 2);
  bf16_t* Vb  = (bf16_t*)carve(M * KD_ * 2);
  bf16_t* Vt  = (bf16_t*)carve(M * KD_ * 2);
  bf16_t* ctx = (bf16_t*)carve(M * D_ * 2);

  // 1) precision conversion (bf16), weights transposed to [N][K]
  cast_kernel<<<(unsigned)(M * D_ / 4 / 256), 256, 0, stream>>>(x, xb, (int)(M * D_));
  tcast_kernel<<<(D_ / 16) * (D_ / 16),  256, 0, stream>>>(wq, wqT, D_, D_);
  tcast_kernel<<<(D_ / 16) * (KD_ / 16), 256, 0, stream>>>(wk, wkT, D_, KD_);
  tcast_kernel<<<(D_ / 16) * (KD_ / 16), 256, 0, stream>>>(wv, wvT, D_, KD_);
  tcast_kernel<<<(D_ / 16) * (D_ / 16),  256, 0, stream>>>(wo, woT, D_, D_);

  // 2) QKV projections (WMMA + TDM-staged LDS tiles)
  gemm_kernel<false><<<dim3((unsigned)(M / 256), D_ / 128),  256, 0, stream>>>(xb, wqT, bq, Qb, (int)M, D_,  D_);
  gemm_kernel<false><<<dim3((unsigned)(M / 256), KD_ / 128), 256, 0, stream>>>(xb, wkT, bk, Kb, (int)M, KD_, D_);
  gemm_kernel<false><<<dim3((unsigned)(M / 256), KD_ / 128), 256, 0, stream>>>(xb, wvT, bv, Vb, (int)M, KD_, D_);

  // 3) V -> [b,g][64][S] for contiguous PV B-fragments
  tbf_kernel<<<dim3((KD_ / 16) * (S_ / 16), B_), 256, 0, stream>>>(Vb, Vt, S_, KD_);

  // 4) flash attention (WMMA QK^T + online softmax + WMMA PV)
  attn_kernel<<<(B_ * NH_ * (S_ / 16)) / 8, 256, 0, stream>>>(Qb, Kb, Vt, ctx);

  // 5) output projection (WMMA, f32 out)
  gemm_kernel<true><<<dim3((unsigned)(M / 256), D_ / 128), 256, 0, stream>>>(ctx, woT, bo, d_out, (int)M, D_, D_);
}
